// EGNNNormConv_72164040507920
// MI455X (gfx1250) — compile-verified, matrix operand order checked by reading, not run on hardware
//
#include <hip/hip_runtime.h>
#include <hip/hip_bf16.h>
#include <stdint.h>

// ---------------------------------------------------------------------------
// EGNN layer for MI455X (gfx1250, wave32, WMMA + Tensor Data Mover).
// All 128-wide GEMMs run on v_wmma_f32_16x16x32_bf16 (fp32 accumulate).
// Weight fragments are DMA'd to LDS with TENSOR_LOAD_TO_LDS per block phase.
// ---------------------------------------------------------------------------

typedef __attribute__((ext_vector_type(16))) __bf16 v16bf;
typedef __attribute__((ext_vector_type(8)))  float  v8f;
typedef unsigned int v4u __attribute__((ext_vector_type(4)));
typedef int          v8i __attribute__((ext_vector_type(8)));
typedef int          v4i __attribute__((ext_vector_type(4)));

union FragA { v16bf v; unsigned u[8]; };

#define TS  66   // LDS A-tile row stride in dwords (132 bf16: 128 data + 4 pad)
#define WPB 4    // waves per block (128 threads)

__device__ __forceinline__ unsigned f2bf_pair(float lo, float hi) {
  unsigned ul = __float_as_uint(lo), uh = __float_as_uint(hi);
  ul = (ul + 0x7FFFu + ((ul >> 16) & 1u)) >> 16;
  uh = (uh + 0x7FFFu + ((uh >> 16) & 1u)) >> 16;
  return (ul & 0xFFFFu) | (uh << 16);
}
__device__ __forceinline__ unsigned short f2bf(float x) {
  unsigned u = __float_as_uint(x);
  u = (u + 0x7FFFu + ((u >> 16) & 1u)) >> 16;
  return (unsigned short)u;
}
__device__ __forceinline__ float silu_f(float x) { return x / (1.0f + __expf(-x)); }
// 16-bit A/B fragment K-pair base per ISA 7.12.2 (VGPR v, lane-half h)
__device__ __forceinline__ int kbase(int v, int h) {
  return ((v & 4) << 2) + ((v & 3) << 1) + (h << 3);
}

// TDM: DMA one 32 KB weight-fragment block (8192 dwords) global -> LDS.
// D# per ISA 8.3/8.4: 1-D tile, data_size=4B, tile_dim0=tensor_dim0=8192.
__device__ __forceinline__ void tdm_fetch_w(const unsigned* gsrc, unsigned* lds_dst) {
  unsigned long long ga = (unsigned long long)(uintptr_t)gsrc;
  v4u g0;
  g0[0] = 1u;                                   // count=1 (valid descriptor)
  g0[1] = (unsigned)(uintptr_t)lds_dst;         // lds_addr (LDS byte offset)
  g0[2] = (unsigned)ga;                         // global_addr[31:0]
  g0[3] = (unsigned)(ga >> 32) | 0x80000000u;   // global_addr[56:32] | type=2
  v8i g1;
  g1[0] = (int)0x20000u;                        // data_size=2 (4 bytes)
  g1[1] = (int)(8192u << 16);                   // tensor_dim0[15:0]
  g1[2] = (int)(1u << 16);                      // tensor_dim0[31:16]=0, tensor_dim1=1
  g1[3] = (int)(8192u << 16);                   // tile_dim0 = 8192
  g1[4] = 0;                                    // tile_dim1/2 unused
  g1[5] = (int)8192u;                           // tensor_dim0_stride[31:0]
  g1[6] = (int)(8192u << 16);                   // tensor_dim1_stride[15:0]
  g1[7] = 0;
  v4i g2 = {0, 0, 0, 0}, g3 = {0, 0, 0, 0};
#if __clang_major__ >= 23
  v8i g4 = {0, 0, 0, 0, 0, 0, 0, 0};
  __builtin_amdgcn_tensor_load_to_lds(g0, g1, g2, g3, g4, 0);
#else
  __builtin_amdgcn_tensor_load_to_lds(g0, g1, g2, g3, 0);
#endif
}

// ---------------------------------------------------------------------------
// Weight prep: pack fp32 weights into bf16 B-matrix fragments.
// Per-matrix dword index: ((nt*4+kt)*8 + v)*32 + lane  (bank-conflict-free
// LDS reads: lane is the fastest axis). 8192 dwords (32 KB) per matrix.
// Matrices: 0=W1a 1=W1b 2=W2 3=W5 4=W3a 5=W3b 6=W4.
// Params (fp32): [w1c, b1, b2, b5, w6, b3, b4] each 128.
// ---------------------------------------------------------------------------
__global__ void egnn_prep_kernel(const float* __restrict__ W1, const float* __restrict__ b1,
                                 const float* __restrict__ W2, const float* __restrict__ b2,
                                 const float* __restrict__ W3, const float* __restrict__ b3,
                                 const float* __restrict__ W4, const float* __restrict__ b4,
                                 const float* __restrict__ W5, const float* __restrict__ b5,
                                 const float* __restrict__ W6,
                                 unsigned* __restrict__ pw, float* __restrict__ params) {
  int t = blockIdx.x * blockDim.x + threadIdx.x;
  if (t < 7 * 8192) {
    int mat = t >> 13, r = t & 8191;
    int lane = r & 31, v = (r >> 5) & 7, kt = (r >> 8) & 3, nt = (r >> 10) & 7;
    int K = kt * 32 + kbase(v, lane >> 4);
    int N = nt * 16 + (lane & 15);
    const float* S; int ro = 0;
    switch (mat) {
      case 0: S = W1; ro = 0;   break;
      case 1: S = W1; ro = 128; break;
      case 2: S = W2;           break;
      case 3: S = W5;           break;
      case 4: S = W3; ro = 0;   break;
      case 5: S = W3; ro = 128; break;
      default: S = W4;          break;
    }
    pw[t] = f2bf_pair(S[(ro + K) * 128 + N], S[(ro + K + 1) * 128 + N]);
  } else {
    int t2 = t - 7 * 8192;
    if (t2 < 7 * 128) {
      int p = t2 >> 7, i = t2 & 127;
      float val;
      switch (p) {
        case 0: val = W1[256 * 128 + i]; break;  // radial row of W1
        case 1: val = b1[i]; break;
        case 2: val = b2[i]; break;
        case 3: val = b5[i]; break;
        case 4: val = W6[i]; break;
        case 5: val = b3[i]; break;
        default: val = b4[i]; break;
      }
      params[t2] = val;
    }
  }
}

// ---------------------------------------------------------------------------
// Edge kernel: 4 waves/block, 16 edges/wave (M=16).
//   msg = silu(silu([h_s,h_d,r]W1+b1)W2+b2);  cw = silu(msg W5+b5)·w6
//   atomic-scatter msg -> h_neigh[dst], cw*x_diff -> x_neigh[dst], 1 -> deg.
// Weight matrices are staged into LDS (32 KB buffer) by TDM, one per phase.
// ---------------------------------------------------------------------------
__global__ __launch_bounds__(128)
void egnn_edge_kernel(const float* __restrict__ node_feat,
                      const float* __restrict__ coord,
                      const int* __restrict__ src,
                      const int* __restrict__ dst,
                      const unsigned* __restrict__ pw,
                      const float* __restrict__ params,
                      float* __restrict__ h_neigh,
                      float* __restrict__ x_neigh,
                      float* __restrict__ deg,
                      long nedges) {
  __shared__ unsigned s_B[8192];              // current weight matrix (32 KB)
  __shared__ unsigned s_tile[WPB][16 * TS];
  __shared__ float s_rad[WPB][16];
  __shared__ float s_xd[WPB][16][3];
  __shared__ int   s_si[WPB][16];
  __shared__ int   s_di[WPB][16];
  __shared__ int   s_ok[WPB][16];

  const int w    = threadIdx.x >> 5;
  const int lane = threadIdx.x & 31;
  const int half = lane >> 4;
  const int mrow = lane & 15;
  const long ebase = (long)blockIdx.x * (WPB * 16) + (long)w * 16;

  const float* w1c = params;
  const float* b1  = params + 128;
  const float* b2  = params + 256;
  const float* b5  = params + 384;
  const float* w6  = params + 512;

  // --- per-edge geometry on lanes 0..15 ---
  if (lane < 16) {
    long e = ebase + lane;
    int ok = (e < nedges);
    int si = 0, di = 0;
    float rad = 0.f, x0 = 0.f, x1 = 0.f, x2 = 0.f;
    if (ok) {
      si = src[e]; di = dst[e];
      x0 = coord[3 * si + 0] - coord[3 * di + 0];
      x1 = coord[3 * si + 1] - coord[3 * di + 1];
      x2 = coord[3 * si + 2] - coord[3 * di + 2];
      rad = x0 * x0 + x1 * x1 + x2 * x2;
      float inv = 1.0f / (sqrtf(rad + 1e-6f) + 1e-6f);
      x0 *= inv; x1 *= inv; x2 *= inv;
    }
    s_si[w][lane] = si; s_di[w][lane] = di; s_ok[w][lane] = ok;
    s_rad[w][lane] = rad;
    s_xd[w][lane][0] = x0; s_xd[w][lane][1] = x1; s_xd[w][lane][2] = x2;
  }
  asm volatile("s_wait_dscnt 0" ::: "memory");

  auto fetchB = [&](int mat) {
    __syncthreads();                          // all waves done with previous B
    if (w == 0) {
      tdm_fetch_w(pw + (size_t)mat * 8192, s_B);
      __builtin_amdgcn_s_wait_tensorcnt((short)0);
    }
    __syncthreads();                          // B visible to all waves
  };
  auto stage = [&](const float* base, bool useDst) {
    for (int i = 0; i < 32; ++i) {
      int r = i >> 1;
      int k2 = ((i & 1) << 5) + lane;         // dword (pair) column 0..63
      int row = useDst ? s_di[w][r] : s_si[w][r];
      const float2 f2 = *(const float2*)(base + (long)row * 128 + k2 * 2);
      s_tile[w][r * TS + k2] = f2bf_pair(f2.x, f2.y);
    }
  };
  auto buildA = [&](FragA* fr) {
#pragma unroll
    for (int kt = 0; kt < 4; ++kt)
#pragma unroll
      for (int v = 0; v < 8; ++v) {
        int K = kt * 32 + kbase(v, half);
        fr[kt].u[v] = s_tile[w][mrow * TS + (K >> 1)];
      }
  };
  // acc += A(16x128) * B_lds[:, nt] ; B frags conflict-free from LDS
  auto gemm_acc = [&](const FragA* a, int nt, v8f acc) -> v8f {
#pragma unroll
    for (int kt = 0; kt < 4; ++kt) {
      FragA bf;
#pragma unroll
      for (int v = 0; v < 8; ++v)
        bf.u[v] = s_B[((nt * 4 + kt) * 8 + v) * 32 + lane];
      acc = __builtin_amdgcn_wmma_f32_16x16x32_bf16(false, a[kt].v, false, bf.v,
                                                    (short)0, acc, false, false);
    }
    return acc;
  };

  FragA a_src[4], a_dst[4], a_h[4];

  stage(node_feat, false);
  asm volatile("s_wait_dscnt 0" ::: "memory");
  buildA(a_src);
  asm volatile("s_wait_dscnt 0" ::: "memory");
  stage(node_feat, true);
  asm volatile("s_wait_dscnt 0" ::: "memory");
  buildA(a_dst);

  unsigned short* stile16 = (unsigned short*)&s_tile[w][0];

  // ---- layer 1 pass A: acc = b1 + radial*w1c + h_src W1a ----
  v8f acc_all[8];
  fetchB(0);
#pragma unroll
  for (int nt = 0; nt < 8; ++nt) {
    float bval = b1[nt * 16 + mrow];
    float wc  = w1c[nt * 16 + mrow];
#pragma unroll
    for (int v = 0; v < 8; ++v)
      acc_all[nt][v] = bval + s_rad[w][v + half * 8] * wc;
    acc_all[nt] = gemm_acc(a_src, nt, acc_all[nt]);
  }
  // ---- layer 1 pass B: acc += h_dst W1b ; silu -> tile (bf16) ----
  fetchB(1);
#pragma unroll
  for (int nt = 0; nt < 8; ++nt) {
    acc_all[nt] = gemm_acc(a_dst, nt, acc_all[nt]);
#pragma unroll
    for (int v = 0; v < 8; ++v) {
      int M = v + half * 8;
      stile16[M * (2 * TS) + nt * 16 + mrow] = f2bf(silu_f(acc_all[nt][v]));
    }
  }
  asm volatile("s_wait_dscnt 0" ::: "memory");
  buildA(a_h);

  // ---- layer 2: msg = silu(. W2 + b2); scatter to h_neigh; keep bf16 in tile
  fetchB(2);
  for (int nt = 0; nt < 8; ++nt) {
    float bval = b2[nt * 16 + mrow];
    v8f acc;
#pragma unroll
    for (int v = 0; v < 8; ++v) acc[v] = bval;
    acc = gemm_acc(a_h, nt, acc);
#pragma unroll
    for (int v = 0; v < 8; ++v) {
      int M = v + half * 8;
      float x = silu_f(acc[v]);
      if (s_ok[w][M])
        atomicAdd(&h_neigh[(long)s_di[w][M] * 128 + nt * 16 + mrow], x);
      stile16[M * (2 * TS) + nt * 16 + mrow] = f2bf(x);
    }
  }
  asm volatile("s_wait_dscnt 0" ::: "memory");
  buildA(a_h);

  // ---- layer 5 + W6: coord weight per edge, then coord scatter ----
  fetchB(3);
  float part[8];
#pragma unroll
  for (int v = 0; v < 8; ++v) part[v] = 0.f;
  for (int nt = 0; nt < 8; ++nt) {
    float bval = b5[nt * 16 + mrow];
    v8f acc;
#pragma unroll
    for (int v = 0; v < 8; ++v) acc[v] = bval;
    acc = gemm_acc(a_h, nt, acc);
    float w6c = w6[nt * 16 + mrow];
#pragma unroll
    for (int v = 0; v < 8; ++v) part[v] += silu_f(acc[v]) * w6c;
  }
#pragma unroll
  for (int v = 0; v < 8; ++v) {
    float s = part[v];
    s += __shfl_xor(s, 1, 16);
    s += __shfl_xor(s, 2, 16);
    s += __shfl_xor(s, 4, 16);
    s += __shfl_xor(s, 8, 16);
    part[v] = s;
  }
  if (mrow == 0) {
#pragma unroll
    for (int v = 0; v < 8; ++v) {
      int M = v + half * 8;
      if (s_ok[w][M]) {
        int di = s_di[w][M];
        float cw = part[v];
        atomicAdd(&x_neigh[3 * di + 0], cw * s_xd[w][M][0]);
        atomicAdd(&x_neigh[3 * di + 1], cw * s_xd[w][M][1]);
        atomicAdd(&x_neigh[3 * di + 2], cw * s_xd[w][M][2]);
        atomicAdd(&deg[di], 1.0f);
      }
    }
  }
}

// ---------------------------------------------------------------------------
// Node kernel: h = silu([h, h_neigh] W3 + b3) W4 + b4 ; x = coord + x_neigh/deg
// plus graph mean/var stat accumulation. Same TDM-staged weight scheme.
// ---------------------------------------------------------------------------
__global__ __launch_bounds__(128)
void egnn_node_kernel(const float* __restrict__ node_feat,
                      const float* __restrict__ coord,
                      const unsigned* __restrict__ pw,
                      const float* __restrict__ params,
                      const float* __restrict__ h_neigh,
                      const float* __restrict__ x_neigh,
                      const float* __restrict__ deg,
                      float* __restrict__ out_h,
                      float* __restrict__ x_tmp,
                      float* __restrict__ stats,
                      int nnodes) {
  __shared__ unsigned s_B[8192];
  __shared__ unsigned s_tile[WPB][16 * TS];
  const int w    = threadIdx.x >> 5;
  const int lane = threadIdx.x & 31;
  const int half = lane >> 4;
  const int mrow = lane & 15;
  const int nbase = blockIdx.x * (WPB * 16) + w * 16;
  const float* b3 = params + 640;
  const float* b4 = params + 768;

  auto fetchB = [&](int mat) {
    __syncthreads();
    if (w == 0) {
      tdm_fetch_w(pw + (size_t)mat * 8192, s_B);
      __builtin_amdgcn_s_wait_tensorcnt((short)0);
    }
    __syncthreads();
  };
  auto stage = [&](const float* base) {
    for (int i = 0; i < 32; ++i) {
      int r = i >> 1;
      int k2 = ((i & 1) << 5) + lane;
      int n = nbase + r; if (n >= nnodes) n = 0;
      const float2 f2 = *(const float2*)(base + (long)n * 128 + k2 * 2);
      s_tile[w][r * TS + k2] = f2bf_pair(f2.x, f2.y);
    }
  };
  auto buildA = [&](FragA* fr) {
#pragma unroll
    for (int kt = 0; kt < 4; ++kt)
#pragma unroll
      for (int v = 0; v < 8; ++v) {
        int K = kt * 32 + kbase(v, half);
        fr[kt].u[v] = s_tile[w][mrow * TS + (K >> 1)];
      }
  };
  auto gemm_acc = [&](const FragA* a, int nt, v8f acc) -> v8f {
#pragma unroll
    for (int kt = 0; kt < 4; ++kt) {
      FragA bf;
#pragma unroll
      for (int v = 0; v < 8; ++v)
        bf.u[v] = s_B[((nt * 4 + kt) * 8 + v) * 32 + lane];
      acc = __builtin_amdgcn_wmma_f32_16x16x32_bf16(false, a[kt].v, false, bf.v,
                                                    (short)0, acc, false, false);
    }
    return acc;
  };

  FragA a1[4], a2[4], ah[4];
  stage(node_feat);
  asm volatile("s_wait_dscnt 0" ::: "memory");
  buildA(a1);
  asm volatile("s_wait_dscnt 0" ::: "memory");
  stage(h_neigh);
  asm volatile("s_wait_dscnt 0" ::: "memory");
  buildA(a2);

  unsigned short* stile16 = (unsigned short*)&s_tile[w][0];

  // layer 3 (two TDM phases: W3a then W3b), 8 live accumulators
  v8f acc_all[8];
  fetchB(4);
#pragma unroll
  for (int nt = 0; nt < 8; ++nt) {
    float bval = b3[nt * 16 + mrow];
#pragma unroll
    for (int v = 0; v < 8; ++v) acc_all[nt][v] = bval;
    acc_all[nt] = gemm_acc(a1, nt, acc_all[nt]);
  }
  fetchB(5);
#pragma unroll
  for (int nt = 0; nt < 8; ++nt) {
    acc_all[nt] = gemm_acc(a2, nt, acc_all[nt]);
#pragma unroll
    for (int v = 0; v < 8; ++v) {
      int M = v + half * 8;
      stile16[M * (2 * TS) + nt * 16 + mrow] = f2bf(silu_f(acc_all[nt][v]));
    }
  }
  asm volatile("s_wait_dscnt 0" ::: "memory");
  buildA(ah);

  // layer 4 -> d_out feature block
  fetchB(6);
  for (int nt = 0; nt < 8; ++nt) {
    float bval = b4[nt * 16 + mrow];
    v8f acc;
#pragma unroll
    for (int v = 0; v < 8; ++v) acc[v] = bval;
    acc = gemm_acc(ah, nt, acc);
#pragma unroll
    for (int v = 0; v < 8; ++v) {
      int M = v + half * 8;
      int n = nbase + M;
      if (n < nnodes) out_h[(long)n * 128 + nt * 16 + mrow] = acc[v];
    }
  }

  // coordinates + mean/var stats
  float sx[3] = {0.f, 0.f, 0.f}, sq[3] = {0.f, 0.f, 0.f};
  if (lane < 16) {
    int n = nbase + lane;
    if (n < nnodes) {
      float dg = deg[n]; if (dg < 1.0f) dg = 1.0f;
#pragma unroll
      for (int c = 0; c < 3; ++c) {
        float xv = coord[3 * n + c] + x_neigh[3 * n + c] / dg;
        x_tmp[3 * n + c] = xv;
        sx[c] = xv; sq[c] = xv * xv;
      }
    }
  }
#pragma unroll
  for (int c = 0; c < 3; ++c) {
    for (int off = 16; off > 0; off >>= 1) {
      sx[c] += __shfl_xor(sx[c], off, 32);
      sq[c] += __shfl_xor(sq[c], off, 32);
    }
  }
  if (lane == 0) {
#pragma unroll
    for (int c = 0; c < 3; ++c) {
      atomicAdd(&stats[c], sx[c]);
      atomicAdd(&stats[3 + c], sq[c]);
    }
  }
}

__global__ void egnn_finalize_kernel(const float* __restrict__ x_tmp,
                                     const float* __restrict__ stats,
                                     float* __restrict__ out_x, int nnodes) {
  int i = blockIdx.x * blockDim.x + threadIdx.x;
  if (i < nnodes * 3) {
    int c = i % 3;
    float inv_n = 1.0f / (float)nnodes;
    float mean = stats[c] * inv_n;
    float var  = stats[3 + c] * inv_n - mean * mean;
    out_x[i] = (x_tmp[i] - mean) / (sqrtf(var + 1e-6f) + 1e-6f);
  }
}

// ---------------------------------------------------------------------------
extern "C" void kernel_launch(void* const* d_in, const int* in_sizes, int n_in,
                              void* d_out, int out_size, void* d_ws, size_t ws_size,
                              hipStream_t stream) {
  const float* node_feat = (const float*)d_in[0];
  const float* coord     = (const float*)d_in[1];
  const int*   src       = (const int*)d_in[2];
  const int*   dst       = (const int*)d_in[3];
  const float* W1 = (const float*)d_in[4];  const float* b1 = (const float*)d_in[5];
  const float* W2 = (const float*)d_in[6];  const float* b2 = (const float*)d_in[7];
  const float* W3 = (const float*)d_in[8];  const float* b3 = (const float*)d_in[9];
  const float* W4 = (const float*)d_in[10]; const float* b4 = (const float*)d_in[11];
  const float* W5 = (const float*)d_in[12]; const float* b5 = (const float*)d_in[13];
  const float* W6 = (const float*)d_in[14];

  const int  nnodes = in_sizes[0] / 128;
  const long nedges = (long)in_sizes[2];

  unsigned* pw     = (unsigned*)d_ws;                      // 7*8192 dwords packed bf16
  float*    params = (float*)d_ws + 7 * 8192;              // 7*128 fp32
  float*    h_neigh = params + 7 * 128;                    // nnodes*128
  float*    x_neigh = h_neigh + (size_t)nnodes * 128;      // nnodes*3
  float*    degv    = x_neigh + (size_t)nnodes * 3;        // nnodes
  float*    x_tmp   = degv + nnodes;                       // nnodes*3
  float*    stats   = x_tmp + (size_t)nnodes * 3;          // 8

  size_t zbytes = (size_t)((stats + 8) - h_neigh) * sizeof(float);
  hipMemsetAsync((void*)h_neigh, 0, zbytes, stream);

  int prep_threads = 7 * 8192 + 7 * 128;
  egnn_prep_kernel<<<(prep_threads + 255) / 256, 256, 0, stream>>>(
      W1, b1, W2, b2, W3, b3, W4, b4, W5, b5, W6, pw, params);

  long eblocks = (nedges + WPB * 16 - 1) / (WPB * 16);
  egnn_edge_kernel<<<(unsigned)eblocks, WPB * 32, 0, stream>>>(
      node_feat, coord, src, dst, pw, params, h_neigh, x_neigh, degv, nedges);

  int nblocks = (nnodes + WPB * 16 - 1) / (WPB * 16);
  egnn_node_kernel<<<nblocks, WPB * 32, 0, stream>>>(
      node_feat, coord, pw, params, h_neigh, x_neigh, degv,
      (float*)d_out, x_tmp, stats, nnodes);

  egnn_finalize_kernel<<<(nnodes * 3 + 255) / 256, 256, 0, stream>>>(
      x_tmp, stats, (float*)d_out + (size_t)nnodes * 128, nnodes);
}
